// MFAWrapper_4114578670419
// MI455X (gfx1250) — matrix operand order repaired and checked by reading, not compiled, not run on hardware
//
#include <hip/hip_runtime.h>
#include <hip/hip_bf16.h>

// ---------------- problem constants ----------------
#define B_N    2048
#define DFT    784
#define K_N    50
#define L_N    6
#define SEC3   (3 * DFT)          // 2352
#define KD     2400               // padded K-dim (75 * 32)
#define NT     (K_N * 32)         // 1600 columns (32 per component)
#define LOG2PI 1.8378770664093453f

typedef __attribute__((ext_vector_type(16))) _Float16 v16h;
typedef __attribute__((ext_vector_type(8)))  _Float16 v8h;
typedef __attribute__((ext_vector_type(8)))  float    v8f;

// upper-triangle enumeration (i<=j) and its packed-lower index j*(j+1)/2+i
static constexpr int TRI_I[21] = {0,0,0,0,0,0, 1,1,1,1,1, 2,2,2,2, 3,3,3, 4,4, 5};
static constexpr int TRI_J[21] = {0,1,2,3,4,5, 1,2,3,4,5, 2,3,4,5, 3,4,5, 4,5, 5};
static constexpr int TRI_L[21] = {0,1,3,6,10,15, 2,4,7,11,16, 5,8,12,17, 9,13,18, 14,19, 20};

// ---------------- kernel 0: build left matrix Lm[B][KD] (f16) ----------------
// sections: [0,784) = Jf, [784,1568) = Jf*X, [1568,2352) = Jf*X^2, rest 0
__global__ void mfa_build_left(const float* __restrict__ X, const int* __restrict__ J,
                               _Float16* __restrict__ Lm) {
    size_t idx = (size_t)blockIdx.x * blockDim.x + threadIdx.x;
    if (idx >= (size_t)B_N * KD) return;
    int b = (int)(idx / KD);
    int d = (int)(idx % KD);
    float val = 0.0f;
    if (d < SEC3) {
        int s  = d / DFT;
        int dd = d - s * DFT;
        if (J[(size_t)b * DFT + dd] == 1) {
            float x = X[(size_t)b * DFT + dd];
            val = (s == 0) ? 1.0f : ((s == 1) ? x : x * x);
        }
    }
    Lm[idx] = (_Float16)val;
}

// ---------------- kernel 1: build right matrix Rt[NT][KD] (f16, N-major) ----------------
// per component k, 32 columns:
//  c in [0,21):  sec0 = invD*A_i*A_j                         (P triangle)
//  c in [21,27): sec0 = -invD*A_i*MU, sec1 = invD*A_i        (q_i)
//  c == 27:      sec0 = invD*MU^2, sec1 = -2*invD*MU, sec2 = invD   (quad1)
//  c == 28:      sec0 = log_D                                (logdet term)
//  c == 29:      sec0 = 1                                    (n_obs)
//  c in {30,31}: 0
__global__ void mfa_build_right(const float* __restrict__ MU, const float* __restrict__ A,
                                const float* __restrict__ logD, _Float16* __restrict__ Rt) {
    size_t idx = (size_t)blockIdx.x * blockDim.x + threadIdx.x;
    if (idx >= (size_t)NT * KD) return;
    int n = (int)(idx / KD);
    int d = (int)(idx % KD);
    int k = n >> 5;
    int c = n & 31;
    float val = 0.0f;
    if (d < SEC3 && c < 30) {
        int s  = d / DFT;
        int dd = d - s * DFT;
        size_t base = (size_t)k * DFT + dd;
        float ld   = logD[base];
        float invd = __expf(-ld);
        float mu   = MU[base];
        const float* a = A + base * L_N;
        if (c < 21) {
            if (s == 0) val = invd * a[TRI_I[c]] * a[TRI_J[c]];
        } else if (c < 27) {
            int i = c - 21;
            val = (s == 0) ? -invd * a[i] * mu : ((s == 1) ? invd * a[i] : 0.0f);
        } else if (c == 27) {
            val = (s == 0) ? invd * mu * mu : ((s == 1) ? -2.0f * invd * mu : invd);
        } else if (c == 28) {
            val = (s == 0) ? ld : 0.0f;
        } else { // c == 29
            val = (s == 0) ? 1.0f : 0.0f;
        }
    }
    Rt[idx] = (_Float16)val;
}

// ---------------- kernel 2: WMMA GEMM  S[B][NT] = Lm[B][KD] * Rt[NT][KD]^T ----------------
// one wave per 32x32 output tile, 2x2 accumulators, f16 16x16x32 WMMA
static __device__ inline v16h ld_frag(const _Float16* p0, const _Float16* p1) {
    union { v16h v; v8h h[2]; } u;
    u.h[0] = *(const v8h*)p0;
    u.h[1] = *(const v8h*)p1;
    return u.v;
}

__global__ __launch_bounds__(32) void mfa_gemm(const _Float16* __restrict__ Lm,
                                               const _Float16* __restrict__ Rt,
                                               float* __restrict__ S) {
    const int lane = threadIdx.x;
    const int m0   = blockIdx.x * 32;
    const int n0   = blockIdx.y * 32;
    const int hi   = lane >> 4;      // 0 or 1
    const int l15  = lane & 15;

    // A fragment (16-bit 16x32): lane<16 holds M=l15, K runs {0..7,16..23};
    // lanes>=16 hold the K runs shifted by 8.  Row-major Lm -> two b128 loads.
    const _Float16* aB0 = Lm + (size_t)(m0 + l15) * KD + hi * 8;
    const _Float16* aB1 = aB0 + (size_t)16 * KD;
    // B fragment (16-bit 32x16): lane<16 holds N=l15, K runs {0..7,16..23} (N-major Rt);
    // lanes>=16 hold the K runs shifted by 8.  Two b128 loads.
    const _Float16* bB0 = Rt + (size_t)(n0 + l15) * KD + hi * 8;
    const _Float16* bB1 = bB0 + (size_t)16 * KD;

    v8f acc00 = {}, acc01 = {}, acc10 = {}, acc11 = {};

    for (int kk = 0; kk < KD; kk += 32) {
        if (kk + 32 < KD) {
            __builtin_prefetch(aB0 + kk + 32, 0, 0);
            __builtin_prefetch(bB0 + kk + 32, 0, 0);
        }
        v16h a0 = ld_frag(aB0 + kk, aB0 + kk + 16);
        v16h a1 = ld_frag(aB1 + kk, aB1 + kk + 16);
        v16h b0 = ld_frag(bB0 + kk, bB0 + kk + 16);
        v16h b1 = ld_frag(bB1 + kk, bB1 + kk + 16);
        acc00 = __builtin_amdgcn_wmma_f32_16x16x32_f16(false, a0, false, b0, (short)0, acc00, false, false);
        acc01 = __builtin_amdgcn_wmma_f32_16x16x32_f16(false, a0, false, b1, (short)0, acc01, false, false);
        acc10 = __builtin_amdgcn_wmma_f32_16x16x32_f16(false, a1, false, b0, (short)0, acc10, false, false);
        acc11 = __builtin_amdgcn_wmma_f32_16x16x32_f16(false, a1, false, b1, (short)0, acc11, false, false);
    }

    // C/D layout: lanes 0-15 VGPR r -> (M=r, N=lane); lanes 16-31 -> (M=r+8, N=lane-16)
    const int ro = hi * 8;
#pragma unroll
    for (int r = 0; r < 8; r++) {
        S[(size_t)(m0 + r + ro)      * NT + n0 + l15]      = acc00[r];
        S[(size_t)(m0 + r + ro)      * NT + n0 + 16 + l15] = acc01[r];
        S[(size_t)(m0 + 16 + r + ro) * NT + n0 + l15]      = acc10[r];
        S[(size_t)(m0 + 16 + r + ro) * NT + n0 + 16 + l15] = acc11[r];
    }
}

// ---------------- kernel 3: per-sample component selection + latent posterior ----------------
#define CIDX(i, j) ((i) * ((i) + 1) / 2 + (j))   // packed lower, i >= j

__global__ void mfa_select(const float* __restrict__ S, const float* __restrict__ PI,
                           int* __restrict__ csel, float* __restrict__ mzout,
                           float* __restrict__ Lzout, float* __restrict__ outPw) {
    int b = blockIdx.x * blockDim.x + threadIdx.x;
    if (b >= B_N) return;

    // log_softmax normalizer over PI_logits
    float mx = -INFINITY;
    for (int k = 0; k < K_N; k++) mx = fmaxf(mx, PI[k]);
    float se = 0.0f;
    for (int k = 0; k < K_N; k++) se += __expf(PI[k] - mx);
    const float lse = mx + __logf(se);

    float best = -INFINITY;
    int   bc   = 0;
    float bC[21], by[6];

    const float* row = S + (size_t)b * NT;
    for (int k = 0; k < K_N; k++) {
        const float* p = row + k * 32;
        float Pm[21];
#pragma unroll
        for (int c = 0; c < 21; c++) Pm[TRI_L[c]] = p[c];
#pragma unroll
        for (int i = 0; i < 6; i++) Pm[CIDX(i, i)] += 1.0f;   // + I

        // Cholesky (packed lower)
        float C[21];
#pragma unroll
        for (int i = 0; i < 6; i++) {
#pragma unroll
            for (int j = 0; j <= i; j++) {
                float s = Pm[CIDX(i, j)];
#pragma unroll
                for (int t = 0; t < j; t++) s -= C[CIDX(i, t)] * C[CIDX(j, t)];
                C[CIDX(i, j)] = (i == j) ? sqrtf(s) : s / C[CIDX(j, j)];
            }
        }
        // forward solve L y = q
        float y[6];
#pragma unroll
        for (int i = 0; i < 6; i++) {
            float s = p[21 + i];
#pragma unroll
            for (int t = 0; t < i; t++) s -= C[CIDX(i, t)] * y[t];
            y[i] = s / C[CIDX(i, i)];
        }
        float quad = p[27];
#pragma unroll
        for (int i = 0; i < 6; i++) quad -= y[i] * y[i];
        float logdet = p[28];
#pragma unroll
        for (int i = 0; i < 6; i++) logdet += 2.0f * __logf(C[CIDX(i, i)]);
        float ll = -0.5f * (quad + logdet + p[29] * LOG2PI) + (PI[k] - lse);
        if (ll > best) {
            best = ll; bc = k;
#pragma unroll
            for (int t = 0; t < 21; t++) bC[t] = C[t];
#pragma unroll
            for (int t = 0; t < 6; t++) by[t] = y[t];
        }
    }

    // m_z: back solve L^T m = y
    float m[6];
#pragma unroll
    for (int i = 5; i >= 0; i--) {
        float s = by[i];
#pragma unroll
        for (int t = i + 1; t < 6; t++) s -= bC[CIDX(t, i)] * m[t];
        m[i] = s / bC[CIDX(i, i)];
    }
    // Linv (lower)
    float Li[21];
#pragma unroll
    for (int j = 0; j < 6; j++) {
        Li[CIDX(j, j)] = 1.0f / bC[CIDX(j, j)];
#pragma unroll
        for (int i = j + 1; i < 6; i++) {
            float s = 0.0f;
#pragma unroll
            for (int t = j; t < i; t++) s += bC[CIDX(i, t)] * Li[CIDX(t, j)];
            Li[CIDX(i, j)] = -s / bC[CIDX(i, i)];
        }
    }
    // cov = L^-T L^-1 (packed lower, i >= j)
    float cov[21];
#pragma unroll
    for (int i = 0; i < 6; i++) {
#pragma unroll
        for (int j = 0; j <= i; j++) {
            float s = 0.0f;
#pragma unroll
            for (int t = i; t < 6; t++) s += Li[CIDX(t, i)] * Li[CIDX(t, j)];
            cov[CIDX(i, j)] = s;
        }
    }
    // Lz = chol(cov)
    float Cz[21];
#pragma unroll
    for (int i = 0; i < 6; i++) {
#pragma unroll
        for (int j = 0; j <= i; j++) {
            float s = cov[CIDX(i, j)];
#pragma unroll
            for (int t = 0; t < j; t++) s -= Cz[CIDX(i, t)] * Cz[CIDX(j, t)];
            Cz[CIDX(i, j)] = (i == j) ? sqrtf(s) : s / Cz[CIDX(j, j)];
        }
    }

    csel[b]  = bc;
    outPw[b] = 1.0f;
#pragma unroll
    for (int i = 0; i < 6; i++) mzout[(size_t)b * 6 + i] = m[i];
#pragma unroll
    for (int i = 0; i < 6; i++)
#pragma unroll
        for (int j = 0; j < 6; j++)
            Lzout[(size_t)b * 36 + i * 6 + j] = (j <= i) ? Cz[CIDX(i, j)] : 0.0f;
}

// ---------------- kernel 4: reconstruction M, A_out, D_out ----------------
__global__ void mfa_recon(const float* __restrict__ MU, const float* __restrict__ A,
                          const float* __restrict__ logD, const int* __restrict__ csel,
                          const float* __restrict__ mz, const float* __restrict__ Lz,
                          float* __restrict__ oM, float* __restrict__ oA,
                          float* __restrict__ oD) {
    __shared__ float sm[6];
    __shared__ float sLz[36];
    __shared__ int   sc;
    const int b   = blockIdx.y;
    const int tid = threadIdx.x;
    if (tid == 0) sc = csel[b];
    if (tid < 6)  sm[tid]  = mz[(size_t)b * 6 + tid];
    if (tid < 36) sLz[tid] = Lz[(size_t)b * 36 + tid];
    __syncthreads();
    const int d = blockIdx.x * blockDim.x + tid;
    if (d >= DFT) return;
    const int c = sc;
    const size_t base = (size_t)c * DFT + d;
    const float* a = A + base * L_N;
    float av[6];
#pragma unroll
    for (int i = 0; i < 6; i++) av[i] = a[i];
    float macc = MU[base];
#pragma unroll
    for (int i = 0; i < 6; i++) macc += av[i] * sm[i];
    const size_t od = (size_t)b * DFT + d;
    oM[od] = macc;
#pragma unroll
    for (int j = 0; j < 6; j++) {
        float s = 0.0f;
#pragma unroll
        for (int i = j; i < 6; i++) s += av[i] * sLz[i * 6 + j];
        oA[od * L_N + j] = s;
    }
    oD[od] = __expf(logD[base]);
}

// ---------------- host launcher ----------------
extern "C" void kernel_launch(void* const* d_in, const int* in_sizes, int n_in,
                              void* d_out, int out_size, void* d_ws, size_t ws_size,
                              hipStream_t stream) {
    const float* X    = (const float*)d_in[0];
    const int*   J    = (const int*)d_in[1];
    const float* MU   = (const float*)d_in[2];
    const float* Afac = (const float*)d_in[3];
    const float* logD = (const float*)d_in[4];
    const float* PI   = (const float*)d_in[5];

    char* ws = (char*)d_ws;
    size_t off = 0;
    _Float16* Lm = (_Float16*)(ws + off); off += (size_t)B_N * KD * sizeof(_Float16);   // 9,830,400
    _Float16* Rt = (_Float16*)(ws + off); off += (size_t)NT * KD * sizeof(_Float16);    // 7,680,000
    float*    S  = (float*)(ws + off);    off += (size_t)B_N * NT * sizeof(float);      // 13,107,200
    int*   csel  = (int*)(ws + off);      off += (size_t)B_N * sizeof(int) + 4096;
    float* mz    = (float*)(ws + off);    off += (size_t)B_N * 6 * sizeof(float);
    float* Lz    = (float*)(ws + off);    off += (size_t)B_N * 36 * sizeof(float);
    (void)ws_size; (void)in_sizes; (void)n_in; (void)out_size;

    float* out = (float*)d_out;
    float* oPw = out;                                 // [B,1]
    float* oM  = out + B_N;                           // [B,DFT]
    float* oA  = oM + (size_t)B_N * DFT;              // [B,DFT,L]
    float* oD  = oA + (size_t)B_N * DFT * L_N;        // [B,DFT]

    {   // build left (f16)
        size_t tot = (size_t)B_N * KD;
        mfa_build_left<<<dim3((unsigned)((tot + 255) / 256)), dim3(256), 0, stream>>>(X, J, Lm);
    }
    {   // build right (f16, N-major)
        size_t tot = (size_t)NT * KD;
        mfa_build_right<<<dim3((unsigned)((tot + 255) / 256)), dim3(256), 0, stream>>>(MU, Afac, logD, Rt);
    }
    {   // WMMA GEMM: 64 x 50 tiles of 32x32, one wave each
        mfa_gemm<<<dim3(B_N / 32, NT / 32), dim3(32), 0, stream>>>(Lm, Rt, S);
    }
    {   // per-sample selection + latent posterior
        mfa_select<<<dim3(B_N / 256), dim3(256), 0, stream>>>(S, PI, csel, mz, Lz, oPw);
    }
    {   // reconstruction
        mfa_recon<<<dim3((DFT + 255) / 256, B_N), dim3(256), 0, stream>>>(MU, Afac, logD, csel, mz, Lz, oM, oA, oD);
    }
}
